// FactorizedSelfAttention_62878321214288
// MI455X (gfx1250) — matrix operand order repaired and checked by reading, not compiled, not run on hardware
//
#include <hip/hip_runtime.h>
#include <hip/hip_bf16.h>

typedef __attribute__((ext_vector_type(16))) _Float16 v16h;
typedef __attribute__((ext_vector_type(8)))  float    v8f;
typedef __attribute__((ext_vector_type(4)))  unsigned int u32x4;
typedef __attribute__((ext_vector_type(8)))  int      i32x8;
typedef __attribute__((ext_vector_type(4)))  int      i32x4;

#define WH (128*128)

static __device__ inline v8f wmma16(v16h a, v16h b, v8f c) {
  return __builtin_amdgcn_wmma_f32_16x16x32_f16(
      /*neg_a=*/false, a, /*neg_b=*/false, b,
      /*c_mod=*/(short)0, c, /*reuse_a=*/false, /*reuse_b=*/false);
}

// A-tile (16x32 f16, M x K) from row-major [m][ld] buffer.
static __device__ inline v16h load_a_rm(const _Float16* p, int ld, int m0, int k0, int lane) {
  const int hi = lane >> 4, m = m0 + (lane & 15);
  v16h a;
#pragma unroll
  for (int j = 0; j < 16; ++j) {
    const int g = j >> 1, lo = j & 1;
    const int k = 16*hi + j;
    a[j] = p[m * ld + k0 + k];
  }
  return a;
}

// B-tile (32x16, K x N) from N-major buffer Bt[n][ldk] == B[k][n].
static __device__ inline v16h load_b_nk(const _Float16* p, int ldk, int n0, int k0, int lane) {
  const int hi = lane >> 4, n = n0 + (lane & 15);
  v16h b;
#pragma unroll
  for (int j = 0; j < 16; ++j) b[j] = p[n * ldk + k0 + 16*hi + j];
  return b;
}

// B-tile from K-major (natural) buffer B[k][ldn].
static __device__ inline v16h load_b_kn(const _Float16* p, int ldn, int k0, int n0, int lane) {
  const int hi = lane >> 4, n = n0 + (lane & 15);
  v16h b;
#pragma unroll
  for (int j = 0; j < 16; ++j) b[j] = p[(k0 + 16*hi + j) * ldn + n];
  return b;
}

// ---------------- Stage 1: fused QKV projection ----------------
// Per block: one batch b, one 64-position slab; x slab staged once in LDS (f16),
// 40 output tiles (10 channel-tiles x 4 pos-tiles) over 8 waves, 4 WMMA k-steps each.
// Q,K stored natural [b][32][WH]; V stored as Vl[b][w][ch][h] (contiguous per (b,w)).
__global__ __launch_bounds__(256) void proj_kernel(
    const float* __restrict__ x,
    const float* __restrict__ wq, const float* __restrict__ bq,
    const float* __restrict__ wk, const float* __restrict__ bk,
    const float* __restrict__ wv, const float* __restrict__ bv,
    _Float16* __restrict__ qk, _Float16* __restrict__ vl)
{
  __shared__ _Float16 Xs[128 * 64];              // Xs[c][p], 16 KB
  const int tid = threadIdx.x, lane = tid & 31, wave = tid >> 5;
  const int hi = lane >> 4, ln = lane & 15;
  const int slab = blockIdx.x & 255, b = blockIdx.x >> 8;
  const int pos0 = slab * 64;
  const float* xb = x + (size_t)b * 128 * WH;

  for (int i = tid; i < 2048; i += 256) {        // 128 rows x 16 float4 each
    const int c = i >> 4, p = (i & 15) * 4;
    const float4 v = *(const float4*)(xb + (size_t)c * WH + pos0 + p);
    Xs[c*64 + p + 0] = (_Float16)v.x;
    Xs[c*64 + p + 1] = (_Float16)v.y;
    Xs[c*64 + p + 2] = (_Float16)v.z;
    Xs[c*64 + p + 3] = (_Float16)v.w;
  }
  __syncthreads();

  for (int s = 0; s < 5; ++s) {
    const int t  = wave * 5 + s;                 // 0..39
    const int mt = t % 10, nt = t / 10;
    const int co = mt * 16 + ln;
    const float* wrow = (co < 16) ? (wq + co * 128)
                       : (co < 32) ? (wk + (co - 16) * 128)
                                   : (wv + (co - 32) * 128);
    v8f acc = {};
#pragma unroll
    for (int kc = 0; kc < 4; ++kc) {
      v16h a;
#pragma unroll
      for (int j = 0; j < 16; ++j) {
        const int g = j >> 1, lo = j & 1;
        const int k = 16*hi + j;
        a[j] = (_Float16)wrow[kc*32 + k];
      }
      const v16h bm = load_b_kn(Xs, 64, kc*32, nt*16, lane);
      acc = wmma16(a, bm, acc);
    }
#pragma unroll
    for (int r = 0; r < 8; ++r) {
      const int cor = mt*16 + r + 8*hi;
      const int pos = pos0 + nt*16 + ln;
      const float bias = (cor < 16) ? bq[cor] : (cor < 32 ? bk[cor-16] : bv[cor-32]);
      const _Float16 val = (_Float16)(acc[r] + bias);
      if (cor < 32) {
        qk[((size_t)b * 32 + cor) * WH + pos] = val;
      } else {
        const int cv = cor - 32, w = pos >> 7, hh = pos & 127;
        vl[(((size_t)b * 128 + w) * 128 + cv) * 128 + hh] = val;
      }
    }
  }
}

// ---------------- Stage 2: row attention, one (b,w) per block ----------------
__global__ __launch_bounds__(256) void row_attn_kernel(
    const _Float16* __restrict__ qk, const _Float16* __restrict__ vl,
    _Float16* __restrict__ ofl)                  // Ofl[b][h][w][c]
{
  __shared__ _Float16 Vs[128*128];               // Vs[h'][c] = V[b,h',w,c] (B natural K-major)
  __shared__ _Float16 Ab[128*128];               // scores -> probs
  const int bw = blockIdx.x, b = bw >> 7, w = bw & 127;
  const int tid = threadIdx.x, lane = tid & 31, wave = tid >> 5;
  const int hi = lane >> 4, ln = lane & 15;
  const _Float16* Qg = qk + (size_t)b * 32 * WH;
  const _Float16* Kg = Qg + (size_t)16 * WH;
  const _Float16* Vg = vl + ((size_t)b * 128 + w) * 16384;

  {                                              // contiguous 32 KB stage: b128 in, b128 to LDS
    const uint4* src = (const uint4*)Vg;
    uint4* dst = (uint4*)Vs;
    for (int i = tid; i < 2048; i += 256) dst[i] = src[i];
  }

  // Q A-tile: Q_row[m][e], flat = m*16+e; e = j (hi==0) -> 16 contiguous f16 in global;
  // the hi==1 half (k>=16) is the zero pad (Ce=16 -> K=32).
  v16h aq = {};
  if (hi == 0) {
    const int m = wave * 16 + ln;
    const int flat0 = m * 16;
    const _Float16* qp = Qg + (size_t)(flat0 >> 7) * WH + w * 128 + (flat0 & 127);
#pragma unroll
    for (int t = 0; t < 16; ++t) aq[t] = qp[t];
  }

  for (int nj = 0; nj < 8; ++nj) {               // S = Q K^T (single padded k-step)
    v16h kb = {};
    if (hi == 0) {                               // k = j = e < 16; hi==1 half is zero pad
      const int flat0 = (nj * 16 + ln) * 16;     // K_row[n][e]: 16 contiguous f16
      const _Float16* kp = Kg + (size_t)(flat0 >> 7) * WH + w * 128 + (flat0 & 127);
#pragma unroll
      for (int j = 0; j < 16; ++j) kb[j] = kp[j];
    }
    v8f s = {};
    s = wmma16(aq, kb, s);
#pragma unroll
    for (int r = 0; r < 8; ++r)
      Ab[(wave*16 + r + 8*hi) * 128 + nj*16 + ln] = (_Float16)s[r];
  }
  __syncthreads();

  if (tid < 128) {                               // softmax over each row of Ab
    _Float16* rowp = Ab + tid * 128;
    float mx = -1e30f;
    for (int j = 0; j < 128; ++j) { const float v = (float)rowp[j]; mx = v > mx ? v : mx; }
    float sum = 0.f;
    for (int j = 0; j < 128; ++j) { const float e = __expf((float)rowp[j] - mx); sum += e; rowp[j] = (_Float16)e; }
    const float inv = 1.f / sum;
    for (int j = 0; j < 128; ++j) rowp[j] = (_Float16)((float)rowp[j] * inv);
  }
  __syncthreads();

  _Float16* ob = ofl + (size_t)b * 128 * 16384;  // [h][w][c]
  for (int nj = 0; nj < 8; ++nj) {               // O = A @ V_row
    v8f acc = {};
#pragma unroll
    for (int kc = 0; kc < 4; ++kc) {
      const v16h a  = load_a_rm(Ab, 128, wave*16, kc*32, lane);
      const v16h bb = load_b_kn(Vs, 128, kc*32, nj*16, lane);
      acc = wmma16(a, bb, acc);
    }
#pragma unroll
    for (int r = 0; r < 8; ++r) {                // coalesced u16 stores (c varies with lane)
      const int h = wave*16 + r + 8*hi, c = nj*16 + ln;
      ob[((size_t)h * 128 + w) * 128 + c] = (_Float16)acc[r];
    }
  }
}

// ---------------- Stage 3: column attention + residual, one (b,h) per block ----------------
__global__ __launch_bounds__(256) void col_attn_kernel(
    const _Float16* __restrict__ ofl, const float* __restrict__ x,
    const float* __restrict__ gamma, float* __restrict__ out)
{
  __shared__ _Float16 Mb[128*128];               // Mb[w][c]: contiguous blob of Ofl[b][h]
  __shared__ _Float16 Ab[128*128];
  const int bh = blockIdx.x, b = bh >> 7, h = bh & 127;
  const int tid = threadIdx.x, lane = tid & 31, wave = tid >> 5;
  const int hi = lane >> 4, ln = lane & 15;
  const _Float16* src = ofl + ((size_t)b * 128 + h) * 16384;

#if __has_builtin(__builtin_amdgcn_tensor_load_to_lds) && __has_builtin(__builtin_amdgcn_s_wait_tensorcnt)
  if (wave == 0) {                               // TDM: 128x128 x 2B tile, global -> LDS
    const unsigned long long ga = (unsigned long long)(uintptr_t)src;
    const unsigned int ldsa = (unsigned int)(uintptr_t)(&Mb[0]);
    u32x4 g0; i32x8 g1 = {}; i32x4 g2 = {}, g3 = {};
    g0[0] = 1u;                                  // count=1, is_restore=0, gather off
    g0[1] = ldsa;                                // lds_addr
    g0[2] = (unsigned int)(ga & 0xffffffffu);    // global_addr[31:0]
    g0[3] = (unsigned int)((ga >> 32) & 0x01ffffffu) | (2u << 30);  // addr[56:32] | type=2
    g1[0] = (1 << 16);                           // data_size = 1 (2 bytes)
    g1[1] = (128 << 16);                         // tensor_dim0 = 128
    g1[2] = (128 << 16);                         // tensor_dim1 = 128
    g1[3] = (128 << 16);                         // tile_dim0 = 128
    g1[4] = 128;                                 // tile_dim1 = 128 (tile_dim2 = 0)
    g1[5] = 128;                                 // tensor_dim0_stride = 128
#if defined(__clang_major__) && __clang_major__ >= 23
    i32x8 g4 = {};
    __builtin_amdgcn_tensor_load_to_lds(g0, g1, g2, g3, g4, 0);
#else
    __builtin_amdgcn_tensor_load_to_lds(g0, g1, g2, g3, 0);
#endif
    __builtin_amdgcn_s_wait_tensorcnt(0);
  }
#else
  {
    const uint4* s4 = (const uint4*)src;
    uint4* d4 = (uint4*)Mb;
    for (int i = tid; i < 2048; i += 256) d4[i] = s4[i];
  }
#endif
  __syncthreads();

  for (int nj = 0; nj < 8; ++nj) {               // S = M M^T
    v8f s = {};
#pragma unroll
    for (int kc = 0; kc < 4; ++kc) {
      const v16h a  = load_a_rm(Mb, 128, wave*16, kc*32, lane);
      const v16h bb = load_b_nk(Mb, 128, nj*16,  kc*32, lane);  // Bt[n][k] = M[n][k]
      s = wmma16(a, bb, s);
    }
#pragma unroll
    for (int r = 0; r < 8; ++r)
      Ab[(wave*16 + r + 8*hi) * 128 + nj*16 + ln] = (_Float16)s[r];
  }
  __syncthreads();

  if (tid < 128) {
    _Float16* rowp = Ab + tid * 128;
    float mx = -1e30f;
    for (int j = 0; j < 128; ++j) { const float v = (float)rowp[j]; mx = v > mx ? v : mx; }
    float sum = 0.f;
    for (int j = 0; j < 128; ++j) { const float e = __expf((float)rowp[j] - mx); sum += e; rowp[j] = (_Float16)e; }
    const float inv = 1.f / sum;
    for (int j = 0; j < 128; ++j) rowp[j] = (_Float16)((float)rowp[j] * inv);
  }
  __syncthreads();

  const float g = gamma[0];
  const float* xb = x   + (size_t)b * 128 * WH + h;
  float*       ob = out + (size_t)b * 128 * WH + h;
  for (int nj = 0; nj < 8; ++nj) {               // O_col = A_col @ M ; out = g*O + x
    v8f acc = {};
#pragma unroll
    for (int kc = 0; kc < 4; ++kc) {
      const v16h a  = load_a_rm(Ab, 128, wave*16, kc*32, lane);
      const v16h bb = load_b_kn(Mb, 128, kc*32,  nj*16, lane);  // B[k=w'][n=c] = Mb[w'][c]
      acc = wmma16(a, bb, acc);
    }
#pragma unroll
    for (int r = 0; r < 8; ++r) {                // out[b, c=n, w=m, h]
      const int w = wave*16 + r + 8*hi, c = nj*16 + ln;
      const size_t off = (size_t)c * WH + (size_t)w * 128;
      ob[off] = g * acc[r] + xb[off];
    }
  }
}

extern "C" void kernel_launch(void* const* d_in, const int* in_sizes, int n_in,
                              void* d_out, int out_size, void* d_ws, size_t ws_size,
                              hipStream_t stream) {
  const float* x     = (const float*)d_in[0];
  const float* wq    = (const float*)d_in[1];
  const float* bq    = (const float*)d_in[2];
  const float* wk    = (const float*)d_in[3];
  const float* bk    = (const float*)d_in[4];
  const float* wv    = (const float*)d_in[5];
  const float* bv    = (const float*)d_in[6];
  const float* gamma = (const float*)d_in[7];
  float* out = (float*)d_out;

  _Float16* qk  = (_Float16*)d_ws;                       // [16][32][16384] f16  (16.8 MB)
  _Float16* vl  = qk  + (size_t)16 * 32  * WH;           // [16][128][128][128] f16 (67 MB)
  _Float16* ofl = vl  + (size_t)16 * 128 * WH;           // [16][128][128][128] f16 (67 MB)

  proj_kernel<<<dim3(16 * 256), dim3(256), 0, stream>>>(
      x, wq, bq, wk, bk, wv, bv, qk, vl);
  row_attn_kernel<<<dim3(16 * 128), dim3(256), 0, stream>>>(qk, vl, ofl);
  col_attn_kernel<<<dim3(16 * 128), dim3(256), 0, stream>>>(ofl, x, gamma, out);
}
